// CausalSelfAttention_79362405695717
// MI455X (gfx1250) — compile-verified, matrix operand order checked by reading, not run on hardware
//
#include <hip/hip_runtime.h>
#include <hip/hip_bf16.h>

// ---------------------------------------------------------------------------
// Causal self-attention forward for MI455X (gfx1250, wave32, WMMA).
// B=2, T=2048, D=1024, H=16, HD=64.
// bf16 end-to-end matrix pipeline, f32 accumulation, b128 fragment loads,
// double-buffered async global->LDS staging (ASYNCcnt software pipeline).
// ---------------------------------------------------------------------------

#define B_BATCH 2
#define T_SEQ   2048
#define D_MODEL 1024
#define N_HEAD  16
#define HDIM    64

typedef __attribute__((ext_vector_type(16))) __bf16 v16bf;
typedef __attribute__((ext_vector_type(8)))  __bf16 v8bf;
typedef __attribute__((ext_vector_type(8)))  float  v8f;

typedef int v4i_ __attribute__((vector_size(16)));
typedef __attribute__((address_space(1))) v4i_ gv4i;   // global 16B vector
typedef __attribute__((address_space(3))) v4i_ lv4i;   // LDS 16B vector

#if __has_builtin(__builtin_amdgcn_global_load_async_to_lds_b128)
#define HAVE_ASYNC_LDS 1
#endif

// Copy 16 bytes global -> LDS (async path if available).
__device__ __forceinline__ void copy16_g2l(void* lds_dst, const void* gsrc) {
#if HAVE_ASYNC_LDS
  __builtin_amdgcn_global_load_async_to_lds_b128((gv4i*)gsrc, (lv4i*)lds_dst,
                                                 0, 0);
#else
  *reinterpret_cast<uint4*>(lds_dst) = *reinterpret_cast<const uint4*>(gsrc);
#endif
}

#if HAVE_ASYNC_LDS && __has_builtin(__builtin_amdgcn_s_wait_asynccnt)
#define ASYNC_WAIT(n) __builtin_amdgcn_s_wait_asynccnt(n)
#else
#define ASYNC_WAIT(n) ((void)0)
#endif

__device__ __forceinline__ v16bf frag_cat(v8bf lo, v8bf hi) {
  return __builtin_shufflevector(lo, hi, 0, 1, 2, 3, 4, 5, 6, 7,
                                 8, 9, 10, 11, 12, 13, 14, 15);
}

// Load one WMMA A/B fragment (K=32) from a row-major 32-element bf16 row.
// Lane's elements: k = half*8 + 0..7  and  k = 16 + half*8 + 0..7.
__device__ __forceinline__ v16bf frag_ld(const __bf16* p, int half) {
  v8bf lo = *reinterpret_cast<const v8bf*>(p + half * 8);
  v8bf hi = *reinterpret_cast<const v8bf*>(p + 16 + half * 8);
  return frag_cat(lo, hi);
}

__device__ __forceinline__ v8f wmma_bf16(v16bf a, v16bf b, v8f c) {
  return __builtin_amdgcn_wmma_f32_16x16x32_bf16(
      false, a, false, b, (short)0, c, false, false);
}

// ---------------------------------------------------------------------------
// Kernel 0: one-time convert/transpose to bf16.
// ---------------------------------------------------------------------------
__global__ void __launch_bounds__(256)
convert_kernel(const float* __restrict__ x, const float* __restrict__ Wqkv,
               const float* __restrict__ Wproj, __bf16* __restrict__ xbf,
               __bf16* __restrict__ Wqkvt, __bf16* __restrict__ Wprojt)
{
  const int idx = blockIdx.x * blockDim.x + threadIdx.x;
  const int NX = B_BATCH * T_SEQ * D_MODEL;            // 4,194,304
  const int NW = D_MODEL * 3 * D_MODEL;                // 3,145,728
  if (idx < NX) {
    xbf[idx] = (__bf16)x[idx];
  } else if (idx < NX + NW) {
    const int e = idx - NX;                            // source (k-major)
    const int k = e / (3 * D_MODEL);
    const int n = e - k * (3 * D_MODEL);
    Wqkvt[(size_t)n * D_MODEL + k] = (__bf16)Wqkv[e];
  } else {
    const int e = idx - NX - NW;
    const int k = e >> 10;
    const int n = e & 1023;
    Wprojt[(size_t)n * D_MODEL + k] = (__bf16)Wproj[e];
  }
}

// ---------------------------------------------------------------------------
// Tiled bf16 GEMM:  C[M,N] = A[M,K] * Bt[N,K]^T + bias[N]
// Block tile 128x256, K-step 32, 8 waves, 64x64 per wave (16 WMMA/K-step).
// Double-buffered async staging: tile kt+1 is in flight while computing kt.
// mode 0: f32 row-major store; mode 1: bf16 scatter into head-major Q/K/V.
// ---------------------------------------------------------------------------
#define BM 128
#define BN 256
#define BK 32

__global__ void __launch_bounds__(256)
gemm_bf16_wmma(const __bf16* __restrict__ A, const __bf16* __restrict__ Bt,
               const float* __restrict__ bias, float* __restrict__ Cout,
               __bf16* __restrict__ Qp, __bf16* __restrict__ Kp,
               __bf16* __restrict__ Vp, int M, int N, int K, int mode)
{
  __shared__ alignas(16) __bf16 sA[2][BM][BK];    // 2 x 8KB
  __shared__ alignas(16) __bf16 sBT[2][BN][BK];   // 2 x 16KB

  const int tid  = threadIdx.x;
  const int lane = tid & 31;
  const int wid  = tid >> 5;
  const int lp   = lane & 15;
  const int half = lane >> 4;

  const int row0 = blockIdx.y * BM;
  const int col0 = blockIdx.x * BN;

  const int wm = wid >> 2;         // 2 row sub-tiles of 64
  const int wn = wid & 3;          // 4 col sub-tiles of 64

  // stage one K-tile into buffer `buf`: 6 async b128 ops per thread
  auto stage = [&](int buf, int k0) {
#pragma unroll
    for (int p = 0; p < 2; ++p) {                  // A: 512 segments
      const int s  = tid + p * 256;
      const int r  = s >> 2;
      const int cs = (s & 3) * 8;
      copy16_g2l(&sA[buf][r][cs], A + (size_t)(row0 + r) * K + k0 + cs);
    }
#pragma unroll
    for (int p = 0; p < 4; ++p) {                  // Bt: 1024 segments
      const int s  = tid + p * 256;
      const int r  = s >> 2;
      const int cs = (s & 3) * 8;
      copy16_g2l(&sBT[buf][r][cs], Bt + (size_t)(col0 + r) * K + k0 + cs);
    }
  };

  v8f acc[2][4] = {};   // [mi pairs share below via two passes? no: see loop]
  // full 4x4 accumulator set:
  v8f acc2[2][4] = {};

  const int kTiles = K / BK;
  stage(0, 0);                                    // prologue

  for (int kt = 0; kt < kTiles; ++kt) {
    const int buf = kt & 1;
    const int k0  = kt * BK;
    if (kt + 1 < kTiles) {
      stage(buf ^ 1, k0 + BK);                    // next tile in flight
      ASYNC_WAIT(6);                              // current tile landed
    } else {
      ASYNC_WAIT(0);
    }
    if (kt + 2 < kTiles) {
      __builtin_prefetch(A  + (size_t)(row0 + (tid >> 1)) * K + k0 + 2 * BK, 0, 3);
      __builtin_prefetch(Bt + (size_t)(col0 + (tid >> 1)) * K + k0 + 2 * BK, 0, 3);
    }
    __syncthreads();

    v16bf af[4];
#pragma unroll
    for (int mi = 0; mi < 4; ++mi)
      af[mi] = frag_ld(&sA[buf][wm * 64 + mi * 16 + lp][0], half);

    v16bf bfr[4];
#pragma unroll
    for (int ni = 0; ni < 4; ++ni)
      bfr[ni] = frag_ld(&sBT[buf][wn * 64 + ni * 16 + lp][0], half);

#pragma unroll
    for (int mi = 0; mi < 2; ++mi)
#pragma unroll
      for (int ni = 0; ni < 4; ++ni) {
        acc[mi][ni]  = wmma_bf16(af[mi],     bfr[ni], acc[mi][ni]);
        acc2[mi][ni] = wmma_bf16(af[mi + 2], bfr[ni], acc2[mi][ni]);
      }

    __syncthreads();
  }

  // store 64x64 per wave
#pragma unroll
  for (int mi = 0; mi < 4; ++mi) {
#pragma unroll
    for (int ni = 0; ni < 4; ++ni) {
      const v8f c = (mi < 2) ? acc[mi][ni] : acc2[mi - 2][ni];
#pragma unroll
      for (int r = 0; r < 8; ++r) {
        const int grow = row0 + wm * 64 + mi * 16 + r + half * 8;
        const int gcol = col0 + wn * 64 + ni * 16 + lp;
        const float val = c[r] + bias[gcol];
        if (mode == 0) {
          Cout[(size_t)grow * N + gcol] = val;
        } else {
          const int sel = gcol >> 10;          // 0=Q 1=K 2=V
          const int rem = gcol & 1023;
          const int h   = rem >> 6;
          const int d   = rem & 63;
          const int bb  = grow >> 11;
          const int t   = grow & (T_SEQ - 1);
          __bf16* dst = (sel == 0) ? Qp : ((sel == 1) ? Kp : Vp);
          dst[(((size_t)(bb * N_HEAD + h)) * T_SEQ + t) * HDIM + d] =
              (__bf16)val;
        }
      }
    }
  }
}

// ---------------------------------------------------------------------------
// RoPE in place on bf16 Q and K (head-major [B*H, T, HD]); f32 math.
// ---------------------------------------------------------------------------
__global__ void __launch_bounds__(256)
rope_kernel(__bf16* __restrict__ Q, __bf16* __restrict__ K, int total)
{
  const int idx = blockIdx.x * blockDim.x + threadIdx.x;
  if (idx >= total) return;
  const int dp  = idx & 31;
  const int bht = idx >> 5;
  const int t   = bht & (T_SEQ - 1);
  const size_t base = (size_t)bht * HDIM;
  const float ang = (float)t * __expf(-(float)dp * 0.28782313663f);
  const float c = __cosf(ang);
  const float s = __sinf(ang);
  const float q1 = (float)Q[base + dp], q2 = (float)Q[base + dp + 32];
  Q[base + dp]      = (__bf16)(q1 * c - q2 * s);
  Q[base + dp + 32] = (__bf16)(q2 * c + q1 * s);
  const float k1 = (float)K[base + dp], k2 = (float)K[base + dp + 32];
  K[base + dp]      = (__bf16)(k1 * c - k2 * s);
  K[base + dp + 32] = (__bf16)(k2 * c + k1 * s);
}

// ---------------------------------------------------------------------------
// Flash attention. One wave per 16-row q-tile of one (b,h); kv chunks of 32.
// K tiles double-buffered through ASYNCcnt; V staged transposed in sync.
// Block: 128 threads = 4 independent waves.
// ---------------------------------------------------------------------------
__global__ void __launch_bounds__(128)
attn_fwd(const __bf16* __restrict__ Q, const __bf16* __restrict__ K,
         const __bf16* __restrict__ V, __bf16* __restrict__ Y)
{
  __shared__ alignas(16) __bf16 sK[4][2][32][HDIM]; // [wave][buf][kv][hd]
  __shared__ alignas(16) __bf16 sVT[4][HDIM][32];   // [wave][hd][kv]
  __shared__ alignas(16) __bf16 sP[4][16][32];      // [wave][q][kv]

  const int lane = threadIdx.x & 31;
  const int wid  = threadIdx.x >> 5;
  const int lp   = lane & 15;
  const int half = lane >> 4;

  const int gw = blockIdx.x * 4 + wid;
  const int qt = gw & (T_SEQ / 16 - 1);
  const int bh = gw >> 7;
  const size_t base = (size_t)bh * T_SEQ * HDIM;

  // per-chunk K stage: 8 async b128 per lane (lane owns one kv row)
  auto stage_k = [&](int buf, int kv0) {
    const __bf16* krow = K + base + (size_t)(kv0 + lane) * HDIM;
#pragma unroll
    for (int c = 0; c < 8; ++c)
      copy16_g2l(&sK[wid][buf][lane][c * 8], krow + c * 8);
  };

  // Q fragments (scale applied to S post-WMMA)
  v16bf qf[2];
#pragma unroll
  for (int f = 0; f < 2; ++f)
    qf[f] = frag_ld(Q + base + (size_t)(qt * 16 + lp) * HDIM + f * 32, half);

  v8f o[4] = {};
  float mrow[8], lrow[8];
#pragma unroll
  for (int r = 0; r < 8; ++r) { mrow[r] = -1.0e30f; lrow[r] = 0.0f; }

  const int jmax = (qt * 16 + 15) >> 5;
  stage_k(0, 0);                                  // prologue

  for (int j = 0; j <= jmax; ++j) {
    const int kv0 = j * 32;
    const int buf = j & 1;

    if (j < jmax) {
      stage_k(buf ^ 1, kv0 + 32);                 // next K tile in flight
      ASYNC_WAIT(8);
    } else {
      ASYNC_WAIT(0);
    }

    // stage V transposed (sync): lane reads its row, scatters columns
    {
      const __bf16* vrow = V + base + (size_t)(kv0 + lane) * HDIM;
#pragma unroll
      for (int c = 0; c < 8; ++c) {
        const v8bf v = *reinterpret_cast<const v8bf*>(vrow + c * 8);
#pragma unroll
        for (int e = 0; e < 8; ++e)
          sVT[wid][c * 8 + e][lane] = v[e];
      }
    }

    // S = Q * K^T (two 16x16 tiles)
    v8f s[2];
#pragma unroll
    for (int tt = 0; tt < 2; ++tt) {
      s[tt] = {};
#pragma unroll
      for (int f = 0; f < 2; ++f) {
        v16bf kb = frag_ld(&sK[wid][buf][tt * 16 + lp][f * 32], half);
        s[tt] = wmma_bf16(qf[f], kb, s[tt]);
      }
#pragma unroll
      for (int r = 0; r < 8; ++r) s[tt][r] *= 0.125f;   // 1/sqrt(64)
    }

    // causal mask on diagonal chunk
    if (j == jmax) {
#pragma unroll
      for (int tt = 0; tt < 2; ++tt)
#pragma unroll
        for (int r = 0; r < 8; ++r)
          if (kv0 + tt * 16 + lp > qt * 16 + r + half * 8)
            s[tt][r] = -1.0e30f;
    }

    // online softmax (row m = r + 8*half spans one 16-lane half)
#pragma unroll
    for (int r = 0; r < 8; ++r) {
      float mx = fmaxf(s[0][r], s[1][r]);
      mx = fmaxf(mx, __shfl_xor(mx, 1, 32));
      mx = fmaxf(mx, __shfl_xor(mx, 2, 32));
      mx = fmaxf(mx, __shfl_xor(mx, 4, 32));
      mx = fmaxf(mx, __shfl_xor(mx, 8, 32));
      const float mnew  = fmaxf(mrow[r], mx);
      const float scale = __expf(mrow[r] - mnew);
      const float p0 = __expf(s[0][r] - mnew);
      const float p1 = __expf(s[1][r] - mnew);
      float sum = p0 + p1;
      sum += __shfl_xor(sum, 1, 32);
      sum += __shfl_xor(sum, 2, 32);
      sum += __shfl_xor(sum, 4, 32);
      sum += __shfl_xor(sum, 8, 32);
      lrow[r] = lrow[r] * scale + sum;
      mrow[r] = mnew;
      s[0][r] = p0;
      s[1][r] = p1;
#pragma unroll
      for (int f = 0; f < 4; ++f) o[f][r] *= scale;
    }

    // P: C-layout -> A-layout via LDS
#pragma unroll
    for (int r = 0; r < 8; ++r) {
      sP[wid][r + half * 8][lp]      = (__bf16)s[0][r];
      sP[wid][r + half * 8][lp + 16] = (__bf16)s[1][r];
    }
    v16bf pf = frag_ld(&sP[wid][lp][0], half);

    // O += P * V
#pragma unroll
    for (int f = 0; f < 4; ++f) {
      v16bf vb = frag_ld(&sVT[wid][f * 16 + lp][0], half);
      o[f] = wmma_bf16(pf, vb, o[f]);
    }
  }

  // finalize, store bf16 Yatt in [B,T,D] layout
  const int b = bh >> 4;
  const int h = bh & 15;
#pragma unroll
  for (int f = 0; f < 4; ++f) {
#pragma unroll
    for (int r = 0; r < 8; ++r) {
      const int t = qt * 16 + r + half * 8;
      Y[((size_t)(b * T_SEQ + t)) * D_MODEL + h * HDIM + f * 16 + lp] =
          (__bf16)(o[f][r] / lrow[r]);
    }
  }
}

// ---------------------------------------------------------------------------
// Host launcher. Workspace (bf16 elements):
//   Qbf | Kbf | Vbf | Ybf | xbf (4,194,304 each)
//   | Wqkvt (3,145,728) | Wprojt (1,048,576)   => 48 MB total
// ---------------------------------------------------------------------------
extern "C" void kernel_launch(void* const* d_in, const int* in_sizes, int n_in,
                              void* d_out, int out_size, void* d_ws,
                              size_t ws_size, hipStream_t stream)
{
  const float* x     = (const float*)d_in[0];
  // d_in[1] = causal mask (computed analytically)
  const float* Wqkv  = (const float*)d_in[2];
  const float* bqkv  = (const float*)d_in[3];
  const float* Wproj = (const float*)d_in[4];
  const float* bproj = (const float*)d_in[5];
  float* out = (float*)d_out;

  const size_t seg = (size_t)B_BATCH * N_HEAD * T_SEQ * HDIM; // 4,194,304
  __bf16* Qbf    = (__bf16*)d_ws;
  __bf16* Kbf    = Qbf + seg;
  __bf16* Vbf    = Kbf + seg;
  __bf16* Ybf    = Vbf + seg;
  __bf16* xbf    = Ybf + seg;
  __bf16* Wqkvt  = xbf + seg;
  __bf16* Wprojt = Wqkvt + (size_t)3 * D_MODEL * D_MODEL;

  const int M = B_BATCH * T_SEQ;   // 4096
  const int K = D_MODEL;           // 1024

  // 0) convert + transpose to bf16
  convert_kernel<<<8388608 / 256, dim3(256), 0, stream>>>(
      x, Wqkv, Wproj, xbf, Wqkvt, Wprojt);

  // 1) QKV projection -> bf16 head-major Q/K/V
  {
    dim3 grid(3 * D_MODEL / BN, M / BM);
    gemm_bf16_wmma<<<grid, dim3(256), 0, stream>>>(
        xbf, Wqkvt, bqkv, nullptr, Qbf, Kbf, Vbf, M, 3 * D_MODEL, K, 1);
  }

  // 2) RoPE in place on Q, K
  {
    const int total = B_BATCH * N_HEAD * T_SEQ * (HDIM / 2);
    rope_kernel<<<total / 256, dim3(256), 0, stream>>>(Qbf, Kbf, total);
  }

  // 3) flash attention -> bf16 Yatt [B,T,D]
  {
    const int waves = B_BATCH * N_HEAD * (T_SEQ / 16);
    attn_fwd<<<waves / 4, dim3(128), 0, stream>>>(Qbf, Kbf, Vbf, Ybf);
  }

  // 4) output projection -> f32 d_out
  {
    dim3 grid(D_MODEL / BN, M / BM);
    gemm_bf16_wmma<<<grid, dim3(256), 0, stream>>>(
        Ybf, Wprojt, bproj, out, nullptr, nullptr, nullptr, M, D_MODEL, K, 0);
  }
}
